// StructFormer_In_Parser_27728308863680
// MI455X (gfx1250) — matrix operand order repaired and checked by reading, not compiled, not run on hardware
//
#include <hip/hip_runtime.h>
#include <math.h>

#define BB   16
#define LL   128
#define DD   512
#define NHH  8
#define HDD  64
#define VV   16000
#define FFD  2048
#define RM   (BB * LL)          // 2048 rows
#define EPSF 1e-5f
#define BIGF 1e9f
#define REARN 4

typedef __bf16 bf16_t;
typedef bf16_t v16bf __attribute__((ext_vector_type(16)));
typedef bf16_t bf16x8 __attribute__((ext_vector_type(8)));
typedef float  v8f   __attribute__((ext_vector_type(8)));

__device__ __forceinline__ long headOff(int z) {
  return (long)(z / NHH) * (LL * DD) + (long)(z % NHH) * HDD;
}

// A lane layout (16x16x32 bf16): e<8 -> k = hl*8+e ; e>=8 -> k = 16+hl*8+(e-8)
__device__ __forceinline__ v16bf loadA(const bf16_t* base) {
  bf16x8 a0 = *reinterpret_cast<const bf16x8*>(base);
  bf16x8 a1 = *reinterpret_cast<const bf16x8*>(base + 16);
  v16bf d;
#pragma unroll
  for (int e = 0; e < 8; ++e) { d[e] = a0[e]; d[e + 8] = a1[e]; }
  return d;
}
// B lane layout: same as A: e<8 -> k = hl*8+e ; e>=8 -> k = 16+hl*8+(e-8)
__device__ __forceinline__ v16bf loadB(const bf16_t* base) {
  bf16x8 b0 = *reinterpret_cast<const bf16x8*>(base);
  bf16x8 b1 = *reinterpret_cast<const bf16x8*>(base + 16);
  v16bf d;
#pragma unroll
  for (int e = 0; e < 8; ++e) { d[e] = b0[e]; d[e + 8] = b1[e]; }
  return d;
}

#define WMMA_BF16(a, b, c) \
  __builtin_amdgcn_wmma_f32_16x16x32_bf16(false, (a), false, (b), (short)0, (c), false, false)

// ---------------------------------------------------------------------------
// bf16 WMMA GEMM, tile-exact (M,N % 64 == 0; K % 64 == 0), branchless hot loop.
// A[m*lda + k], B[n*ldbn + k] both bf16 contiguous in k.
// C f32 (nullable, supports accumulate) and/or Cbf bf16 (nullable).
// ---------------------------------------------------------------------------
struct GemmF {
  const bf16_t* A; const bf16_t* Bm; const float* bias;
  float* C; bf16_t* Cbf;
  int K;
  int lda, ldbn, ldc;
  int modeA, modeB, modeC;          // 1 = attention head view offset
  long long zsA, zsB, zsC;
  float alpha; int act; int accum;
};

__global__ __launch_bounds__(128)
void k_gemm_bf(GemmF p) {
  const int lane = threadIdx.x & 31;
  const int wv   = threadIdx.x >> 5;               // 4 waves -> 64x64 WG tile
  const int tileM = blockIdx.y * 64 + (wv >> 1) * 32;
  const int tileN = blockIdx.x * 64 + (wv & 1) * 32;
  const int z = blockIdx.z;
  const bf16_t* A  = p.A  + (p.modeA ? headOff(z) : (long)(p.zsA * z));
  const bf16_t* Bm = p.Bm + (p.modeB ? headOff(z) : (long)(p.zsB * z));
  const long coff  = (p.modeC ? headOff(z) : (long)(p.zsC * z));
  const int hl = lane >> 4, r = lane & 15;

  const bf16_t* Ar0 = A + (long)(tileM + r) * p.lda + hl * 8;
  const bf16_t* Ar1 = Ar0 + (long)16 * p.lda;
  const bf16_t* Bc0 = Bm + (long)(tileN + r) * p.ldbn + hl * 8;
  const bf16_t* Bc1 = Bc0 + (long)16 * p.ldbn;

  v8f acc[2][2];
#pragma unroll
  for (int sm = 0; sm < 2; ++sm)
#pragma unroll
    for (int sn = 0; sn < 2; ++sn) {
      if (p.accum) {
        const float* c0 = p.C + coff + (long)(tileM + sm * 16 + hl * 8) * p.ldc
                        + tileN + sn * 16 + r;
#pragma unroll
        for (int vx = 0; vx < 8; ++vx) acc[sm][sn][vx] = c0[(long)vx * p.ldc];
      } else {
#pragma unroll
        for (int vx = 0; vx < 8; ++vx) acc[sm][sn][vx] = 0.f;
      }
    }

  for (int k0 = 0; k0 < p.K; k0 += 64) {
#pragma unroll
    for (int u = 0; u < 2; ++u) {                  // 8 WMMA per loop body
      const int kb = k0 + u * 32;
      v16bf a0 = loadA(Ar0 + kb);
      v16bf a1 = loadA(Ar1 + kb);
      v16bf b0 = loadB(Bc0 + kb);
      v16bf b1 = loadB(Bc1 + kb);
      __builtin_prefetch(Bc0 + kb + 128, 0, 1);
      acc[0][0] = WMMA_BF16(a0, b0, acc[0][0]);
      acc[0][1] = WMMA_BF16(a0, b1, acc[0][1]);
      acc[1][0] = WMMA_BF16(a1, b0, acc[1][0]);
      acc[1][1] = WMMA_BF16(a1, b1, acc[1][1]);
    }
  }

#pragma unroll
  for (int sm = 0; sm < 2; ++sm)
#pragma unroll
    for (int sn = 0; sn < 2; ++sn) {
      float b0 = p.bias ? p.bias[tileN + sn * 16 + r] : 0.f;
      const long e0 = coff + (long)(tileM + sm * 16 + hl * 8) * p.ldc
                    + tileN + sn * 16 + r;
#pragma unroll
      for (int vx = 0; vx < 8; ++vx) {
        float val = (acc[sm][sn][vx] + b0) * p.alpha;
        if (p.act == 1) val = (val > 0.f) ? val : 0.01f * val;
        if (p.C)   p.C[e0 + (long)vx * p.ldc] = val;
        if (p.Cbf) p.Cbf[e0 + (long)vx * p.ldc] = (bf16_t)val;
      }
    }
}

// ---------------------------------------------------------------------------
// Conv1d tap GEMM (bf16): M=2048, N=K=512, lda=ldc=512, B per-tap contiguous.
// Shifted rows outside [0,L) read a zero bf16 scratch row (branchless select).
// ---------------------------------------------------------------------------
struct GemmC {
  const bf16_t* A; const bf16_t* Bm; const float* bias; float* C;
  const bf16_t* zrow;
  int shift; int accum;
};

__global__ __launch_bounds__(128)
void k_gemm_conv(GemmC p) {
  const int lane = threadIdx.x & 31;
  const int wv   = threadIdx.x >> 5;
  const int tileM = blockIdx.y * 64 + (wv >> 1) * 32;
  const int tileN = blockIdx.x * 64 + (wv & 1) * 32;
  const int hl = lane >> 4, r = lane & 15;

  const bf16_t* Ar[2];
  const bf16_t* Bc[2];
#pragma unroll
  for (int s = 0; s < 2; ++s) {
    int m = tileM + s * 16 + r;
    int bidx = m >> 7;
    int l = (m & 127) + p.shift;
    const bf16_t* row = ((unsigned)l < (unsigned)LL)
                      ? (p.A + ((long)((bidx << 7) + l)) * DD) : p.zrow;
    Ar[s] = row + hl * 8;
    Bc[s] = p.Bm + (long)(tileN + s * 16 + r) * DD + hl * 8;
  }

  v8f acc[2][2];
#pragma unroll
  for (int sm = 0; sm < 2; ++sm)
#pragma unroll
    for (int sn = 0; sn < 2; ++sn) {
      if (p.accum) {
        const float* c0 = p.C + (long)(tileM + sm * 16 + hl * 8) * DD + tileN + sn * 16 + r;
#pragma unroll
        for (int vx = 0; vx < 8; ++vx) acc[sm][sn][vx] = c0[(long)vx * DD];
      } else {
#pragma unroll
        for (int vx = 0; vx < 8; ++vx) acc[sm][sn][vx] = 0.f;
      }
    }

  for (int k0 = 0; k0 < DD; k0 += 64) {
#pragma unroll
    for (int u = 0; u < 2; ++u) {
      const int kb = k0 + u * 32;
      v16bf a0 = loadA(Ar[0] + kb);
      v16bf a1 = loadA(Ar[1] + kb);
      v16bf b0 = loadB(Bc[0] + kb);
      v16bf b1 = loadB(Bc[1] + kb);
      acc[0][0] = WMMA_BF16(a0, b0, acc[0][0]);
      acc[0][1] = WMMA_BF16(a0, b1, acc[0][1]);
      acc[1][0] = WMMA_BF16(a1, b0, acc[1][0]);
      acc[1][1] = WMMA_BF16(a1, b1, acc[1][1]);
    }
  }

#pragma unroll
  for (int sm = 0; sm < 2; ++sm)
#pragma unroll
    for (int sn = 0; sn < 2; ++sn) {
      float b0 = p.bias ? p.bias[tileN + sn * 16 + r] : 0.f;
      float* c0 = p.C + (long)(tileM + sm * 16 + hl * 8) * DD + tileN + sn * 16 + r;
#pragma unroll
      for (int vx = 0; vx < 8; ++vx) c0[(long)vx * DD] = acc[sm][sn][vx] + b0;
    }
}

// ---------------------------------------------------------------------------
// Conversions / repacks
// ---------------------------------------------------------------------------
__global__ void k_cvt(const float* in, bf16_t* out, int n) {
  int i = blockIdx.x * blockDim.x + threadIdx.x;
  if (i < n) out[i] = (bf16_t)in[i];
}
// (O,I,KW) torch conv weight -> KW contiguous (O,I) bf16 tap matrices
__global__ void k_packw(const float* w, bf16_t* out, int kw, int n) {
  int idx = blockIdx.x * blockDim.x + threadIdx.x;
  if (idx >= n) return;
  int i = idx & (DD - 1);
  int o = (idx >> 9) & (DD - 1);
  int t = idx >> 18;
  out[idx] = (bf16_t)w[((long)o * DD + i) * kw + t];
}
__global__ void k_zerob(bf16_t* p, int n) {
  int i = blockIdx.x * blockDim.x + threadIdx.x;
  if (i < n) p[i] = (bf16_t)0.f;
}
// per-head V transpose: (L,HD) slice of (B,L,D) -> contiguous (HD,L) bf16
__global__ __launch_bounds__(128)
void k_vtrans(const bf16_t* v, bf16_t* vt) {
  int z = blockIdx.x;                 // b*NH + h
  int j = threadIdx.x;                // L
  const bf16_t* src = v + headOff(z) + (long)j * DD;
  bf16_t* dst = vt + (long)z * (HDD * LL) + j;
#pragma unroll 4
  for (int d = 0; d < HDD; ++d) dst[(long)d * LL] = src[d];
}

// ---------------------------------------------------------------------------
// LayerNorm over D=512 (optional affine/tanh); f32 and/or bf16 outputs.
// ---------------------------------------------------------------------------
__global__ __launch_bounds__(128)
void k_ln(const float* x, const float* g, const float* b,
          float* y, bf16_t* ybf, int act) {
  __shared__ float red[128];
  const int row = blockIdx.x, t = threadIdx.x;
  const float* xr = x + (long)row * DD;
  float v[4]; float s = 0.f;
#pragma unroll
  for (int i = 0; i < 4; ++i) { v[i] = xr[t + i * 128]; s += v[i]; }
  red[t] = s; __syncthreads();
  for (int st = 64; st > 0; st >>= 1) { if (t < st) red[t] += red[t + st]; __syncthreads(); }
  float mean = red[0] * (1.f / DD); __syncthreads();
  float q = 0.f;
#pragma unroll
  for (int i = 0; i < 4; ++i) { float d = v[i] - mean; q += d * d; }
  red[t] = q; __syncthreads();
  for (int st = 64; st > 0; st >>= 1) { if (t < st) red[t] += red[t + st]; __syncthreads(); }
  float rs = rsqrtf(red[0] * (1.f / DD) + EPSF);
#pragma unroll
  for (int i = 0; i < 4; ++i) {
    int c = t + i * 128;
    float o = (v[i] - mean) * rs;
    if (g) o = o * g[c] + b[c];
    if (act) o = tanhf(o);
    if (y)   y[(long)row * DD + c] = o;
    if (ybf) ybf[(long)row * DD + c] = (bf16_t)o;
  }
}

// Row dot with a (1,D) weight, + scalar bias
__global__ __launch_bounds__(128)
void k_rowdot(const float* x, const float* w, const float* bias, float* out) {
  __shared__ float red[128];
  const int row = blockIdx.x, t = threadIdx.x;
  const float* xr = x + (long)row * DD;
  float s = 0.f;
#pragma unroll
  for (int i = 0; i < 4; ++i) s += xr[t + i * 128] * w[t + i * 128];
  red[t] = s; __syncthreads();
  for (int st = 64; st > 0; st >>= 1) { if (t < st) red[t] += red[t + st]; __syncthreads(); }
  if (t == 0) out[row] = red[0] + bias[0];
}

// Embedding gather
__global__ __launch_bounds__(128)
void k_embed(const int* x, const float* emb, float* h) {
  const int row = blockIdx.x, t = threadIdx.x;
  long src = (long)x[row] * DD;
#pragma unroll
  for (int i = 0; i < 4; ++i)
    h[(long)row * DD + t + i * 128] = emb[src + t + i * 128];
}

// scores += rel_bias[|i-j|+256]; softmax (mode0) or sigmoid * structural mask
// (mode1, mask = softmax(rel_weight[r]) . {head, head^T} computed inline).
// Output bf16 (A operand of the w@v GEMM).
__global__ __launch_bounds__(128)
void k_score_post(const float* scores, bf16_t* outb, const float* rel_bias,
                  const float* headT, const float* relw, int r, int mode) {
  __shared__ float red[128];
  const int rowg = blockIdx.x;               // n*L + i
  const int n = rowg >> 7, i = rowg & 127, j = threadIdx.x;
  const int brow = n / BB;                   // repeat_interleave quirk
  const int d = (i > j) ? (i - j) : (j - i);
  float sc = scores[(long)rowg * LL + j] + rel_bias[brow * 512 + 256 + d];
  if (mode == 1) {
    int b = n >> 3, hh = n & 7;              // n = b*NH + hh
    float w0 = relw[(r * NHH + hh) * 2 + 0];
    float w1 = relw[(r * NHH + hh) * 2 + 1];
    float mx = fmaxf(w0, w1);
    float e0 = expf(w0 - mx), e1 = expf(w1 - mx);
    float inv = 1.f / (e0 + e1);
    float att = (e0 * inv) * headT[((long)(b * LL + i)) * LL + j]
              + (e1 * inv) * headT[((long)(b * LL + j)) * LL + i];
    float w = 1.f / (1.f + expf(-sc));
    outb[(long)rowg * LL + j] = (bf16_t)(w * att);
  } else {
    red[j] = sc; __syncthreads();
    for (int st = 64; st > 0; st >>= 1) { if (j < st) red[j] = fmaxf(red[j], red[j + st]); __syncthreads(); }
    float mx = red[0]; __syncthreads();
    float e = expf(sc - mx);
    red[j] = e; __syncthreads();
    for (int st = 64; st > 0; st >>= 1) { if (j < st) red[j] += red[j + st]; __syncthreads(); }
    outb[(long)rowg * LL + j] = (bf16_t)(e / red[0]);
  }
}

__global__ void k_maskdist(float* dist) {
  int b = threadIdx.x;
  if (b < BB) dist[b * LL + LL - 1] = BIGF;
}

// per-(b,i) margin row-max (running cummax of height over [i..j])
__global__ void k_margin_rowmax(const float* dist, const float* height, float* rowmax) {
  int gid = blockIdx.x * blockDim.x + threadIdx.x;
  if (gid >= RM) return;
  int b = gid >> 7, i = gid & 127;
  const float* db = dist + b * LL;
  const float* hb = height + b * LL;
  float dpad = (i == 0) ? BIGF : db[i - 1];
  float hm = -BIGF, rmax = 0.f;
  for (int j = i; j < LL; ++j) {
    hm = fmaxf(hm, hb[j]);
    float mg = fminf(fmaxf(dpad - hm, 0.f), fmaxf(db[j] - hm, 0.f));
    if (i == 0 && j == LL - 1) mg = 0.f;
    rmax = fmaxf(rmax, mg);
  }
  rowmax[gid] = rmax;
}

__global__ void k_reduce_max(const float* in, int n, float* out) {
  __shared__ float red[256];
  int t = threadIdx.x;
  float m = -BIGF;
  for (int i = t; i < n; i += 256) m = fmaxf(m, in[i]);
  red[t] = m; __syncthreads();
  for (int st = 128; st > 0; st >>= 1) { if (t < st) red[t] = fmaxf(red[t], red[t + st]); __syncthreads(); }
  if (t == 0) out[0] = red[0];
}

__global__ void k_subs(float* x, const float* s, int n) {
  int i = blockIdx.x * blockDim.x + threadIdx.x;
  if (i < n) x[i] -= s[0];
}

// bml/bmr: reverse/forward cummin + adjacent-difference, row per lane (LDS row)
__global__ __launch_bounds__(64)
void k_block(const float* dist, const float* height, const float* scaler,
             float* bml, float* bmr) {
  __shared__ float buf[64 * LL];             // 32 KB of the 320 KB WGP LDS
  int gid = blockIdx.x * 64 + threadIdx.x;
  if (gid >= RM) return;
  int b = gid >> 7, q = gid & 127;
  float s0 = expf(scaler[0]);
  float hq = height[b * LL + q];
  const float* db = dist + b * LL;
  float* row = buf + threadIdx.x * LL;
  float c = 0.f;
  for (int j = LL - 1; j >= 0; --j) {        // reverse cummin of gamma*tl + tu
    float val = (j >= q) ? 1.f : 1.f / (1.f + expf((db[j] - hq) * s0));
    c = (j == LL - 1) ? val : fminf(c, val);
    row[j] = c;
  }
  float* bm = bml + (long)gid * LL;
  for (int j = 0; j < LL; ++j)
    bm[j] = (j <= q) ? (row[j] - (j ? row[j - 1] : 0.f)) : 0.f;
  for (int j = 0; j < LL; ++j) {             // forward cummin of shifted gamma*tu + tl
    float rin = (j == 0) ? 1.f
              : ((j - 1 >= q) ? 1.f / (1.f + expf((db[j - 1] - hq) * s0)) : 1.f);
    c = (j == 0) ? rin : fminf(c, rin);
    row[j] = c;
  }
  float* br = bmr + (long)gid * LL;
  for (int j = 0; j < LL; ++j)
    br[j] = (j >= q) ? (row[j] - (j < LL - 1 ? row[j + 1] : 0.f)) : 0.f;
}

// e = exp(height*s1) and per-batch prefix sums (Z denominators)
__global__ void k_expheight(const float* height, const float* scaler, float* evec, float* Zp) {
  int b = threadIdx.x;
  if (b >= BB) return;
  float s1 = expf(scaler[1]);
  float acc = 0.f;
  Zp[b * (LL + 1)] = 0.f;
  for (int h = 0; h < LL; ++h) {
    float e = expf(height[b * LL + h] * s1);
    evec[b * LL + h] = e;
    acc += e;
    Zp[b * (LL + 1) + h + 1] = acc;
  }
}

// head[b,l,h] = e[h] * sum_{i<=h} bml_i * suffix_j( bmr_j / (Zp[j+1]-Zp[i]) )
// factored einsum: O(L^2) per (b,l) with LDS f32 atomics instead of O(L^3) GEMM.
__global__ __launch_bounds__(128)
void k_headp(const float* bml, const float* bmr, const float* evec, const float* Zp,
             float* headT) {
  __shared__ float T[LL];
  __shared__ float Rsh[LL];
  __shared__ float Zsh[LL + 1];
  const int bl = blockIdx.x;
  const int b = bl >> 7, l = bl & 127;
  const int t = threadIdx.x;                 // t == i
  T[t] = 0.f;
  Rsh[t] = bmr[(long)bl * LL + t];
  Zsh[t] = Zp[b * (LL + 1) + t];
  if (t == 0) Zsh[LL] = Zp[b * (LL + 1) + LL];
  __syncthreads();
  float Mi = bml[(long)bl * LL + t];
  float Zi = Zsh[t];
  if (Mi != 0.f) {
    float S = 0.f;
    for (int h = LL - 1; h >= t; --h) {
      S += Rsh[h] / (Zsh[h + 1] - Zi);
      atomicAdd(&T[h], Mi * S);              // ds_add_f32
    }
  }
  __syncthreads();
  headT[(long)bl * LL + t] = (t == l) ? 0.f : evec[b * LL + t] * T[t];
}

// ---------------------------------------------------------------------------
extern "C" void kernel_launch(void* const* d_in, const int* in_sizes, int n_in,
                              void* d_out, int out_size, void* d_ws, size_t ws_size,
                              hipStream_t stream) {
  (void)in_sizes; (void)n_in; (void)out_size; (void)ws_size;
  auto fin = [&](int i) { return (const float*)d_in[i]; };
  const int* xtok = (const int*)d_in[0];
  const float* emb = fin(1);
  // index map: 0 x | 1 emb | 2+17i front | 70+17i rear | 138..143 parser w/b
  // 144 d_w 145 d_b 146 d_w2 147 d_b2 | 148 h_w1 149 h_b1 150 h_w2 151 h_b2
  // 152 rel_weight 153 scaler 154 norm_g 155 norm_b 156 out_b

  char* wsb = (char*)d_ws;
  size_t off = 0;
  auto allocf = [&](size_t n) { off = (off + 15) & ~(size_t)15; float* p = (float*)(wsb + off); off += n * 4; return p; };
  auto allocb = [&](size_t n) { off = (off + 15) & ~(size_t)15; bf16_t* p = (bf16_t*)(wsb + off); off += n * 2; return p; };

  // f32 buffers
  float* hbuf   = allocf((size_t)RM * DD);
  float* tmp1   = allocf((size_t)RM * DD);
  float* tmp2   = allocf((size_t)RM * DD);
  float* scores = allocf((size_t)BB * NHH * LL * LL);
  float* convb  = allocf((size_t)RM * DD);
  float* bml    = allocf((size_t)RM * LL);
  float* bmr    = allocf((size_t)RM * LL);
  float* headT  = allocf((size_t)RM * LL);
  float* height = allocf(RM);
  float* dist   = allocf(RM);
  float* evec   = allocf(RM);
  float* Zp     = allocf(BB * (LL + 1));
  float* rowmax = allocf(RM);
  float* gmax   = allocf(64);
  // bf16 activations
  bf16_t* a_bf    = allocb((size_t)RM * DD);
  bf16_t* qb_bf   = allocb((size_t)RM * DD);
  bf16_t* kb_bf   = allocb((size_t)RM * DD);
  bf16_t* vb_bf   = allocb((size_t)RM * DD);
  bf16_t* vt_bf   = allocb((size_t)RM * DD);
  bf16_t* sc_bf   = allocb((size_t)BB * NHH * LL * LL);
  bf16_t* ao_bf   = allocb((size_t)RM * DD);
  bf16_t* ffn_bf  = allocb((size_t)RM * FFD);
  bf16_t* h_bf    = allocb((size_t)RM * DD);
  bf16_t* hp_bf   = allocb((size_t)RM * DD);
  bf16_t* zrowb   = allocb(DD);
  // bf16 weights
  bf16_t* embB = allocb((size_t)VV * DD);
  bf16_t* layW[8][6];
  for (int i = 0; i < 8; ++i) {
    layW[i][0] = allocb((size_t)DD * DD);    // wq
    layW[i][1] = allocb((size_t)DD * DD);    // wk
    layW[i][2] = allocb((size_t)DD * DD);    // wv
    layW[i][3] = allocb((size_t)DD * DD);    // wo
    layW[i][4] = allocb((size_t)FFD * DD);   // w1
    layW[i][5] = allocb((size_t)DD * FFD);   // w2
  }
  bf16_t* packB[3];
  for (int pl = 0; pl < 3; ++pl) packB[pl] = allocb((size_t)9 * DD * DD);
  bf16_t* dwB  = allocb((size_t)2 * DD * DD);
  bf16_t* hw1B = allocb((size_t)DD * DD);

  auto cvt = [&](const float* in, bf16_t* out, size_t n) {
    k_cvt<<<(unsigned)((n + 255) / 256), 256, 0, stream>>>(in, out, (int)n);
  };
  // ---- one-time (per call) weight conversion / repack
  cvt(emb, embB, (size_t)VV * DD);
  for (int i = 0; i < 8; ++i) {
    int base = (i < 4) ? (2 + i * 17) : (70 + (i - 4) * 17);
    cvt(fin(base + 0),  layW[i][0], (size_t)DD * DD);
    cvt(fin(base + 2),  layW[i][1], (size_t)DD * DD);
    cvt(fin(base + 4),  layW[i][2], (size_t)DD * DD);
    cvt(fin(base + 6),  layW[i][3], (size_t)DD * DD);
    cvt(fin(base + 13), layW[i][4], (size_t)FFD * DD);
    cvt(fin(base + 15), layW[i][5], (size_t)DD * FFD);
  }
  for (int pl = 0; pl < 3; ++pl) {
    int n = 9 * DD * DD;
    k_packw<<<(n + 255) / 256, 256, 0, stream>>>(fin(138 + pl * 2), packB[pl], 9, n);
  }
  { int n = 2 * DD * DD;
    k_packw<<<(n + 255) / 256, 256, 0, stream>>>(fin(144), dwB, 2, n); }
  cvt(fin(148), hw1B, (size_t)DD * DD);
  k_zerob<<<2, 256, 0, stream>>>(zrowb, DD);

  auto GF = [&](const bf16_t* A, const bf16_t* Bm, const float* bias,
                float* C, bf16_t* Cbf, int K, int lda, int ldbn, int ldc) {
    GemmF p{}; p.A = A; p.Bm = Bm; p.bias = bias; p.C = C; p.Cbf = Cbf;
    p.K = K; p.lda = lda; p.ldbn = ldbn; p.ldc = ldc; p.alpha = 1.f; return p;
  };
  auto GLf = [&](const GemmF& p, int M, int N, int Z) {
    dim3 grid(N / 64, M / 64, Z);
    k_gemm_bf<<<grid, dim3(128), 0, stream>>>(p);
  };

  // ---- embedding
  k_embed<<<RM, 128, 0, stream>>>(xtok, emb, hbuf);

  // ---- transformer layer (rearIdx < 0: softmax; else sigmoid * structural mask)
  const float* relw = fin(152);
  auto tlayer = [&](int li, int rearIdx) {
    int base = (li < 4) ? (2 + li * 17) : (70 + (li - 4) * 17);
    const float *bq = fin(base + 1), *bk = fin(base + 3), *bv = fin(base + 5);
    const float *bo = fin(base + 7), *rb = fin(base + 8);
    const float *ng = fin(base + 9),  *nb = fin(base + 10);
    const float *fg = fin(base + 11), *fb = fin(base + 12);
    const float *b1 = fin(base + 14), *b2 = fin(base + 16);

    k_ln<<<RM, 128, 0, stream>>>(hbuf, ng, nb, nullptr, a_bf, 0);
    { GemmF p = GF(a_bf, layW[li][0], bq, nullptr, qb_bf, DD, DD, DD, DD);
      p.alpha = 0.125f; GLf(p, RM, DD, 1); }                     // q * HD^-0.5
    { GemmF p = GF(a_bf, layW[li][1], bk, nullptr, kb_bf, DD, DD, DD, DD); GLf(p, RM, DD, 1); }
    { GemmF p = GF(a_bf, layW[li][2], bv, nullptr, vb_bf, DD, DD, DD, DD); GLf(p, RM, DD, 1); }
    k_vtrans<<<BB * NHH, 128, 0, stream>>>(vb_bf, vt_bf);
    // scores = q @ k^T per head (z = b*NH+h), K = 64
    { GemmF p = GF(qb_bf, kb_bf, nullptr, scores, nullptr, HDD, DD, DD, LL);
      p.modeA = 1; p.modeB = 1; p.zsC = (long long)LL * LL; GLf(p, LL, LL, BB * NHH); }
    k_score_post<<<BB * NHH * LL, 128, 0, stream>>>(scores, sc_bf, rb, headT, relw,
                                                    rearIdx, rearIdx >= 0 ? 1 : 0);
    // out = w @ v per head (B = transposed V, contiguous)
    { GemmF p = GF(sc_bf, vt_bf, nullptr, nullptr, ao_bf, LL, LL, LL, DD);
      p.zsA = (long long)LL * LL; p.zsB = (long long)HDD * LL; p.modeC = 1;
      GLf(p, LL, HDD, BB * NHH); }
    // h += out @ wo^T + bo   (residual folded into accumulate)
    { GemmF p = GF(ao_bf, layW[li][3], bo, hbuf, nullptr, DD, DD, DD, DD);
      p.accum = 1; GLf(p, RM, DD, 1); }
    k_ln<<<RM, 128, 0, stream>>>(hbuf, fg, fb, nullptr, a_bf, 0);
    { GemmF p = GF(a_bf, layW[li][4], b1, nullptr, ffn_bf, DD, DD, DD, FFD);
      p.act = 1; GLf(p, RM, FFD, 1); }
    { GemmF p = GF(ffn_bf, layW[li][5], b2, hbuf, nullptr, FFD, FFD, FFD, DD);
      p.accum = 1; GLf(p, RM, DD, 1); }
  };

  for (int i = 0; i < 4; ++i) tlayer(i, -1);

  // ---- parser: 3x [conv1d(k=9) as 9 shift-accumulate WMMA GEMMs -> LN -> tanh]
  cvt(hbuf, h_bf, (size_t)RM * DD);
  const bf16_t* cur = h_bf;
  dim3 cgrid(DD / 64, RM / 64, 1);
  for (int pl = 0; pl < 3; ++pl) {
    const float* cb = fin(139 + pl * 2);
    for (int t = 0; t < 9; ++t) {
      GemmC p{}; p.A = cur; p.Bm = packB[pl] + (size_t)t * DD * DD;
      p.bias = (t == 8) ? cb : nullptr; p.C = convb; p.zrow = zrowb;
      p.shift = t - 4; p.accum = (t > 0);
      k_gemm_conv<<<cgrid, dim3(128), 0, stream>>>(p);
    }
    k_ln<<<RM, 128, 0, stream>>>(convb, nullptr, nullptr, nullptr, hp_bf, 1);
    cur = hp_bf;
  }
  // height head
  { GemmF p = GF(hp_bf, hw1B, fin(149), tmp1, nullptr, DD, DD, DD, DD); GLf(p, RM, DD, 1); }
  k_ln<<<RM, 128, 0, stream>>>(tmp1, nullptr, nullptr, tmp2, nullptr, 1);
  k_rowdot<<<RM, 128, 0, stream>>>(tmp2, fin(150), fin(151), height);
  // distance head: conv k=2 -> taps {0,+1}
  for (int t = 0; t < 2; ++t) {
    GemmC p{}; p.A = hp_bf; p.Bm = dwB + (size_t)t * DD * DD;
    p.bias = (t == 1) ? fin(145) : nullptr; p.C = convb; p.zrow = zrowb;
    p.shift = t; p.accum = (t > 0);
    k_gemm_conv<<<cgrid, dim3(128), 0, stream>>>(p);
  }
  k_ln<<<RM, 128, 0, stream>>>(convb, nullptr, nullptr, tmp2, nullptr, 1);
  k_rowdot<<<RM, 128, 0, stream>>>(tmp2, fin(146), fin(147), dist);
  k_maskdist<<<1, BB, 0, stream>>>(dist);
  // global margin max, dist -= max
  k_margin_rowmax<<<16, 128, 0, stream>>>(dist, height, rowmax);
  k_reduce_max<<<1, 256, 0, stream>>>(rowmax, RM, gmax);
  k_subs<<<(RM + 255) / 256, 256, 0, stream>>>(dist, gmax, RM);

  // ---- structural parse tensors
  const float* scaler = fin(153);
  k_block<<<RM / 64, 64, 0, stream>>>(dist, height, scaler, bml, bmr);
  k_expheight<<<1, 32, 0, stream>>>(height, scaler, evec, Zp);
  k_headp<<<RM, 128, 0, stream>>>(bml, bmr, evec, Zp, headT);

  // ---- rear layers (mask applied inline in k_score_post from headT/rel_weight)
  for (int i = 0; i < 4; ++i) tlayer(4 + i, i);

  // ---- final LN + tied-embedding logits (M=2048, N=16000, K=512)
  k_ln<<<RM, 128, 0, stream>>>(hbuf, fin(154), fin(155), nullptr, a_bf, 0);
  { GemmF p = GF(a_bf, embB, fin(156), (float*)d_out, nullptr, DD, DD, DD, VV);
    GLf(p, RM, VV, 1); }
}
